// GCN_61486751809886
// MI455X (gfx1250) — matrix operand order repaired and checked by reading, not compiled, not run on hardware
//
#include <hip/hip_runtime.h>

// GCN forward for MI455X (gfx1250, wave32).
// GEMMs via v_wmma_f32_16x16x32_f16 with LDS B pre-swizzled into fragment
// layout (all fragment loads are ds_load_b128); edge aggregation is
// wave-per-edge with scalar index loads + coalesced float4 gathers +
// global_atomic_add_f32 scatter (hot buffers are L2-resident: 192MB L2).

typedef __attribute__((ext_vector_type(8)))  _Float16 v8h;
typedef __attribute__((ext_vector_type(16))) _Float16 v16h;
typedef __attribute__((ext_vector_type(8)))  float    v8f;

#define C_HID 128
#define C_OUT 40
#define EPSV  1e-5f

// ---------------- utility kernels ----------------

__global__ __launch_bounds__(256) void k_fill(float* p, float v, int n) {
  int i = blockIdx.x * 256 + threadIdx.x;
  if (i < n) p[i] = v;
}

__global__ __launch_bounds__(256) void k_deg_atomic(const int* __restrict__ dst,
                                                    float* __restrict__ deg, int E) {
  int e = blockIdx.x * 256 + threadIdx.x;
  if (e < E) atomicAdd(&deg[dst[e]], 1.0f);
}

__global__ __launch_bounds__(256) void k_rsqrt_inplace(float* p, int n) {
  int i = blockIdx.x * 256 + threadIdx.x;
  if (i < n) p[i] = rsqrtf(p[i]);
}

// -------- fragment helpers (layouts per CDNA5 ISA 7.12.2) --------
// A (16x32 f16): lane=hi*16+hl holds row hl; halfs 0..7 <-> K=8*hi+j,
//                halfs 8..15 <-> K=16+8*hi+j.
// B (32x16 f16): lane=hi*16+hl holds col hl; half j <-> K=8*hi+j (j<8), K=16+8*hi+(j-8) (j>=8).
// C/D (16x16 f32): VGPR r <-> M=r+8*hi, N=hl.

__device__ __forceinline__ v16h load_frag16(const _Float16* p) {
  v8h lo = *(const v8h*)p;
  v8h hi = *(const v8h*)(p + 8);
  return __builtin_shufflevector(lo, hi, 0, 1, 2, 3, 4, 5, 6, 7,
                                 8, 9, 10, 11, 12, 13, 14, 15);
}

// ---------------- GEMM1: h0 = x @ W1  [N,128]x[128,128] ----------------
// Block: 256 threads (8 waves), 64 rows x 128 cols.
// wave -> rowTile = wave>>1 (16 rows), colHalf = wave&1 (4 col-tiles of 16).

__global__ __launch_bounds__(256) void k_gemm1(const float* __restrict__ x,
                                               const float* __restrict__ W1,
                                               float* __restrict__ h0, int N) {
  __shared__ alignas(16) _Float16 sA[64][136];        // x tile, f16
  __shared__ alignas(16) _Float16 sBf[8][4][32][24];  // W1 in fragment layout
  const int tid = threadIdx.x;
  const int rowBase = blockIdx.x * 64;

  // W1 -> fragment-swizzled LDS: element (k,n) -> sBf[n>>4][k>>5][((k>>3)&1)*16+hl][((k>>1)&8)|(k&7)]
  for (int i = tid; i < 128 * 128; i += 256) {
    int k = i >> 7, n = i & 127;
    int ct = n >> 4, hl = n & 15;
    int kc = k >> 5, hi = (k >> 3) & 1, j = ((k >> 1) & 8) | (k & 7);
    sBf[ct][kc][hi * 16 + hl][j] = (_Float16)W1[i];
  }
  for (int i = tid; i < 64 * 128; i += 256) {
    int r = i >> 7, c = i & 127;
    int row = rowBase + r;
    sA[r][c] = (row < N) ? (_Float16)x[(size_t)row * 128 + c] : (_Float16)0.f;
  }
  __syncthreads();

  const int wave = tid >> 5;
  const int lane = tid & 31;
  const int hl = lane & 15, hi = lane >> 4;
  const int rt = wave >> 1;        // row tile 0..3
  const int ch = wave & 1;         // col half: tiles ch*4 .. ch*4+3

  v8f acc[4] = {};
#pragma unroll
  for (int kc = 0; kc < 4; ++kc) {
    const int k0 = kc * 32;
    v8h alo = *(const v8h*)&sA[rt * 16 + hl][k0 + hi * 8];
    v8h ahi = *(const v8h*)&sA[rt * 16 + hl][k0 + 16 + hi * 8];
    v16h a = __builtin_shufflevector(alo, ahi, 0, 1, 2, 3, 4, 5, 6, 7,
                                     8, 9, 10, 11, 12, 13, 14, 15);
#pragma unroll
    for (int i = 0; i < 4; ++i) {
      v16h b = load_frag16(&sBf[ch * 4 + i][kc][lane][0]);
      acc[i] = __builtin_amdgcn_wmma_f32_16x16x32_f16(false, a, false, b,
                                                      (short)0, acc[i], false, false);
    }
  }
#pragma unroll
  for (int i = 0; i < 4; ++i) {
    int col = (ch * 4 + i) * 16 + hl;
#pragma unroll
    for (int r = 0; r < 8; ++r) {
      int row = rowBase + rt * 16 + hi * 8 + r;
      if (row < N) h0[(size_t)row * 128 + col] = acc[i][r];
    }
  }
}

// ---------------- edge aggregation 1 (wave per edge, float4/lane) ------------

__global__ __launch_bounds__(256) void k_agg1(const int* __restrict__ src,
                                              const int* __restrict__ dst,
                                              const float* __restrict__ dinv,
                                              const float* __restrict__ h0,
                                              float* __restrict__ hagg, int E) {
  int gid = blockIdx.x * 256 + threadIdx.x;
  int e = __builtin_amdgcn_readfirstlane(gid >> 5);  // wave-uniform edge id
  if (e >= E) return;
  int lane = threadIdx.x & 31;
  int s = src[e], d = dst[e];
  float c = dinv[s] * dinv[d];
  float4 v = *(const float4*)(h0 + (size_t)s * 128 + lane * 4);
  float* o = hagg + (size_t)d * 128 + lane * 4;
  atomicAdd(o + 0, v.x * c);
  atomicAdd(o + 1, v.y * c);
  atomicAdd(o + 2, v.z * c);
  atomicAdd(o + 3, v.w * c);
}

// self-loop contribution + bias:  hagg += h0 * dinv^2 + b1
__global__ __launch_bounds__(256) void k_selfbias1(const float* __restrict__ h0,
                                                   const float* __restrict__ dinv,
                                                   const float* __restrict__ b1,
                                                   float* __restrict__ hagg, int N) {
  int idx = blockIdx.x * 256 + threadIdx.x;
  if (idx >= N * 128) return;
  int row = idx >> 7, c = idx & 127;
  float di = dinv[row];
  hagg[idx] += h0[idx] * di * di + b1[c];
}

// ---------------- BatchNorm stats + apply(+ReLU, f16 emit) ----------------

__global__ __launch_bounds__(256) void k_bnstats(const float* __restrict__ hagg,
                                                 float* __restrict__ sums,
                                                 float* __restrict__ sumq, int N) {
  int c = threadIdx.x & 127;
  int r0 = blockIdx.x * 2 + (threadIdx.x >> 7);
  int stride = gridDim.x * 2;
  float s = 0.f, q = 0.f;
  for (int r = r0; r < N; r += stride) {
    float v = hagg[(size_t)r * 128 + c];
    s += v;
    q += v * v;
  }
  atomicAdd(&sums[c], s);
  atomicAdd(&sumq[c], q);
}

__global__ __launch_bounds__(256) void k_bnapply(const float* __restrict__ hagg,
                                                 const float* __restrict__ sums,
                                                 const float* __restrict__ sumq,
                                                 const float* __restrict__ gma,
                                                 const float* __restrict__ bta,
                                                 _Float16* __restrict__ hrelu, int N) {
  int idx = blockIdx.x * 256 + threadIdx.x;
  if (idx >= N * 128) return;
  int c = idx & 127;
  float invN = 1.0f / (float)N;
  float mean = sums[c] * invN;
  float var = sumq[c] * invN - mean * mean;
  float sc = gma[c] * rsqrtf(var + EPSV);
  float y = (hagg[idx] - mean) * sc + bta[c];
  hrelu[idx] = (_Float16)fmaxf(y, 0.f);
}

// ---------------- GEMM2: h2 = hrelu @ W2  [N,128]x[128,40] ----------------
// Block: 96 threads (3 waves), 16 rows x 48 cols (cols 40..47 zero padded).

__global__ __launch_bounds__(96) void k_gemm2(const _Float16* __restrict__ hrelu,
                                              const float* __restrict__ W2,
                                              float* __restrict__ h2, int N) {
  __shared__ alignas(16) _Float16 sA[16][136];
  __shared__ alignas(16) _Float16 sBf[3][4][32][24];
  const int tid = threadIdx.x;
  const int rowBase = blockIdx.x * 16;

  for (int i = tid; i < 128 * 48; i += 96) {
    int k = i / 48, n = i - k * 48;
    int ct = n >> 4, hl = n & 15;
    int kc = k >> 5, hi = (k >> 3) & 1, j = ((k >> 1) & 8) | (k & 7);
    sBf[ct][kc][hi * 16 + hl][j] =
        (n < C_OUT) ? (_Float16)W2[k * C_OUT + n] : (_Float16)0.f;
  }
  for (int i = tid; i < 16 * 128; i += 96) {
    int r = i >> 7, c = i & 127;
    int row = rowBase + r;
    sA[r][c] = (row < N) ? hrelu[(size_t)row * 128 + c] : (_Float16)0.f;
  }
  __syncthreads();

  const int wave = tid >> 5;  // col tile 0..2
  const int lane = tid & 31;
  const int hl = lane & 15, hi = lane >> 4;

  v8f acc = {};
#pragma unroll
  for (int kc = 0; kc < 4; ++kc) {
    const int k0 = kc * 32;
    v8h alo = *(const v8h*)&sA[hl][k0 + hi * 8];
    v8h ahi = *(const v8h*)&sA[hl][k0 + 16 + hi * 8];
    v16h a = __builtin_shufflevector(alo, ahi, 0, 1, 2, 3, 4, 5, 6, 7,
                                     8, 9, 10, 11, 12, 13, 14, 15);
    v16h b = load_frag16(&sBf[wave][kc][lane][0]);
    acc = __builtin_amdgcn_wmma_f32_16x16x32_f16(false, a, false, b,
                                                 (short)0, acc, false, false);
  }
  int col = wave * 16 + hl;
  if (col < C_OUT) {
#pragma unroll
    for (int r = 0; r < 8; ++r) {
      int row = rowBase + hi * 8 + r;
      if (row < N) h2[(size_t)row * C_OUT + col] = acc[r];
    }
  }
}

// ---------------- output init + edge aggregation 2 (40 ch/edge) --------------

__global__ __launch_bounds__(256) void k_outinit(const float* __restrict__ h2,
                                                 const float* __restrict__ dinv,
                                                 const float* __restrict__ b2,
                                                 float* __restrict__ out, int N) {
  int idx = blockIdx.x * 256 + threadIdx.x;
  if (idx >= N * C_OUT) return;
  int row = idx / C_OUT;
  int c = idx - row * C_OUT;
  float di = dinv[row];
  out[idx] = h2[idx] * di * di + b2[c];
}

__global__ __launch_bounds__(256) void k_agg2(const int* __restrict__ src,
                                              const int* __restrict__ dst,
                                              const float* __restrict__ dinv,
                                              const float* __restrict__ h2,
                                              float* __restrict__ out, int E) {
  int gid = blockIdx.x * 256 + threadIdx.x;
  int e = __builtin_amdgcn_readfirstlane(gid >> 5);  // wave-uniform edge id
  if (e >= E) return;
  int lane = threadIdx.x & 31;
  int ch = lane * 2;
  if (ch >= C_OUT) return;
  int s = src[e], d = dst[e];
  float c = dinv[s] * dinv[d];
  float2 v = *(const float2*)(h2 + (size_t)s * C_OUT + ch);
  float* o = out + (size_t)d * C_OUT + ch;
  atomicAdd(o + 0, v.x * c);
  atomicAdd(o + 1, v.y * c);
}

// ---------------- host launcher ----------------

extern "C" void kernel_launch(void* const* d_in, const int* in_sizes, int n_in,
                              void* d_out, int out_size, void* d_ws, size_t ws_size,
                              hipStream_t stream) {
  const float* x   = (const float*)d_in[0];
  const int*   ei  = (const int*)d_in[1];
  const float* W1  = (const float*)d_in[2];
  const float* b1  = (const float*)d_in[3];
  const float* gma = (const float*)d_in[4];
  const float* bta = (const float*)d_in[5];
  const float* W2  = (const float*)d_in[6];
  const float* b2  = (const float*)d_in[7];
  float* out = (float*)d_out;

  const int N = in_sizes[0] / C_HID;  // C_IN == C_HID == 128
  const int E = in_sizes[1] / 2;
  const int* srcp = ei;
  const int* dstp = ei + E;

  char* ws = (char*)d_ws;
  size_t off = 0;
  float* h0   = (float*)(ws + off); off += (size_t)N * C_HID * sizeof(float);
  float* hagg = (float*)(ws + off); off += (size_t)N * C_HID * sizeof(float);
  float* h2   = (float*)(ws + off); off += (size_t)N * C_OUT * sizeof(float);
  float* dinv = (float*)(ws + off); off += (size_t)N * sizeof(float);
  float* sums = (float*)(ws + off); off += 128 * sizeof(float);
  float* sumq = (float*)(ws + off); off += 128 * sizeof(float);
  _Float16* hrelu = (_Float16*)h0;  // alias: h0 is dead before hrelu is written

  const int nElem1 = N * C_HID;  // 6.4M
  const int nElem2 = N * C_OUT;  // 2.0M

  // 1) degree (init 1.0 for self-loop) -> dinv
  k_fill<<<(N + 255) / 256, 256, 0, stream>>>(dinv, 1.0f, N);
  k_deg_atomic<<<(E + 255) / 256, 256, 0, stream>>>(dstp, dinv, E);
  k_rsqrt_inplace<<<(N + 255) / 256, 256, 0, stream>>>(dinv, N);

  // 2) h0 = x @ W1 (WMMA, 64 rows/block)
  k_gemm1<<<(N + 63) / 64, 256, 0, stream>>>(x, W1, h0, N);

  // 3) hagg = scatter-add of edge messages, then self-loop + bias
  k_fill<<<(nElem1 + 255) / 256, 256, 0, stream>>>(hagg, 0.0f, nElem1);
  k_agg1<<<(E * 32 + 255) / 256, 256, 0, stream>>>(srcp, dstp, dinv, h0, hagg, E);
  k_selfbias1<<<(nElem1 + 255) / 256, 256, 0, stream>>>(h0, dinv, b1, hagg, N);

  // 4) BatchNorm (batch stats) + ReLU -> f16
  k_fill<<<1, 256, 0, stream>>>(sums, 0.0f, 256);  // zeros sums[128] + sumq[128]
  k_bnstats<<<512, 256, 0, stream>>>(hagg, sums, sumq, N);
  k_bnapply<<<(nElem1 + 255) / 256, 256, 0, stream>>>(hagg, sums, sumq, gma, bta, hrelu, N);

  // 5) h2 = hrelu @ W2 (WMMA)
  k_gemm2<<<(N + 15) / 16, 96, 0, stream>>>(hrelu, W2, h2, N);

  // 6) out = scatter-add + self-loop + bias
  k_outinit<<<(nElem2 + 255) / 256, 256, 0, stream>>>(h2, dinv, b2, out, N);
  k_agg2<<<(E * 32 + 255) / 256, 256, 0, stream>>>(srcp, dstp, dinv, h2, out, E);
}
